// FOFE_NN_att_1340029796585
// MI455X (gfx1250) — compile-verified, matrix operand order checked by reading, not run on hardware
//
#include <hip/hip_runtime.h>
#include <hip/hip_bf16.h>
#include <stdint.h>

// ---------------------------------------------------------------------------
// Problem constants (from the reference)
// ---------------------------------------------------------------------------
#define BB    8          // batch
#define LQ    64         // query length
#define LL    1024       // doc length
#define DD    128        // embedding dim
#define NH    15         // MAX_LEN-1 window lengths (2..16)
#define NPIX  (NH*BB*LL) // 122880 pixels, ordering pix = (h*B + b)*L + l
#define ROWS  (LL + 32)  // padded FOFE rows (16 zeros each side)
#define ALPHAF 0.9f

typedef __attribute__((ext_vector_type(16))) __bf16          v16bf;
typedef __attribute__((ext_vector_type(16))) unsigned short  v16u16;
typedef __attribute__((ext_vector_type(8)))  float           v8f;

__device__ __forceinline__ unsigned short f2bf_bits(float f) {
  union { float f; unsigned u; } v; v.f = f;
  unsigned r = v.u + 0x7FFFu + ((v.u >> 16) & 1u);   // round-to-nearest-even
  return (unsigned short)(r >> 16);
}
__device__ __forceinline__ float lrelu(float v) { return v > 0.f ? v : 0.1f * v; }

__device__ __forceinline__ v16bf ldfrag(const unsigned short* p) {
  return __builtin_bit_cast(v16bf, *(const v16u16*)p);
}
__device__ __forceinline__ v8f bfwmma(v16bf a, v16bf b, v8f c) {
  return __builtin_amdgcn_wmma_f32_16x16x32_bf16(false, a, false, b,
                                                 (short)0, c, false, false);
}

// ---- CDNA5 async global->LDS staging (ASYNCcnt-tracked DMA) ---------------
// Builtin signature (from hipcc diagnostic): params are int4-vector pointers,
// global (addrspace 1) source and LDS (addrspace 3) destination.
#if defined(__AMDGCN__) && __has_builtin(__builtin_amdgcn_global_load_async_to_lds_b128)
#define HAVE_ASYNC 1
typedef __attribute__((ext_vector_type(4))) int v4i;
typedef __attribute__((address_space(1))) v4i v4i_g;
typedef __attribute__((address_space(3))) v4i v4i_l;
__device__ __forceinline__ void async_copy_b128(const void* g, void* l) {
  __builtin_amdgcn_global_load_async_to_lds_b128(
      (v4i_g*)(uintptr_t)g, (v4i_l*)(uint32_t)(uintptr_t)l, 0, 0);
}
#else
#define HAVE_ASYNC 0
#endif

__device__ __forceinline__ void wait_async() {
#if HAVE_ASYNC
#if __has_builtin(__builtin_amdgcn_s_wait_asynccnt)
  __builtin_amdgcn_s_wait_asynccnt(0);
#else
  asm volatile("s_wait_asynccnt 0x0" ::: "memory");
#endif
#endif
}

// ---------------------------------------------------------------------------
// Generic bf16 WMMA GEMM:  OUT[n, m] = sum_k W[m,k] * IN[n,k]
//   W  : bf16 [M][K] row-major
//   IN : IN_F32_BN ? f32 [N][K] (BN+LeakyReLU fused on load) : bf16 [N][K]
//   OUT: OUT_BF16 ? bf16 (LeakyReLU applied), stride/offset addressable
//                 : f32 [N][M] + per-channel sum/sumsq atomics (for BatchNorm)
// Block = 256 threads (8 waves), tile 128(M) x 128(N), K staged 32 at a time
// through double-buffered LDS (async DMA when available).
// Wave w (4x2): M-sub (w&3)*32 (2 WMMA tiles), N-sub (w>>2)*64 (4 tiles)
//   -> 8 accumulators, 8 v_wmma per K-step per wave.
// ---------------------------------------------------------------------------
#define TILE_M 128
#define TILE_N 128
#define KSTEP  32

template <int IN_F32_BN, int OUT_BF16>
__global__ void __launch_bounds__(256)
gemm_wmma(const unsigned short* __restrict__ Wb,
          const void* __restrict__ INp,
          const float* __restrict__ in_mean,
          const float* __restrict__ in_rstd,
          void* __restrict__ OUTp,
          float* __restrict__ stat_sum,
          float* __restrict__ stat_sq,
          int M, int N, int K, int out_stride, int out_off)
{
  __shared__ __align__(16) unsigned short ldsA[2][TILE_M * KSTEP];
  __shared__ __align__(16) unsigned short ldsB[2][TILE_N * KSTEP];
  __shared__ float lsum[TILE_M];
  __shared__ float lsq[TILE_M];
  (void)M; (void)N;

  const int tid   = threadIdx.x;
  const int nBase = blockIdx.x * TILE_N;
  const int mBase = blockIdx.y * TILE_M;

  if (!OUT_BF16 && tid < TILE_M) { lsum[tid] = 0.f; lsq[tid] = 0.f; }

  // Cooperative staging: thread -> row tid>>1, 16-element half (tid&1)
  const int srow = tid >> 1;          // 0..127
  const int scol = (tid & 1) * 16;    // 0 or 16

  auto stageA = [&](int k0, int buf) {
    const unsigned short* src = Wb + (size_t)(mBase + srow) * K + k0 + scol;
    unsigned short*       dst = &ldsA[buf][srow * KSTEP + scol];
#if HAVE_ASYNC
    async_copy_b128(src, dst);
    async_copy_b128(src + 8, dst + 8);
#else
    *(uint4*)dst       = *(const uint4*)src;
    *(uint4*)(dst + 8) = *(const uint4*)(src + 8);
    if (k0 + KSTEP < K) __builtin_prefetch((const void*)(src + KSTEP), 0, 0);
#endif
  };

  auto stageB = [&](int k0, int buf) {
    unsigned short* dst = &ldsB[buf][srow * KSTEP + scol];
    if (IN_F32_BN) {
      const float* src = (const float*)INp + (size_t)(nBase + srow) * K + k0 + scol;
#pragma unroll
      for (int j = 0; j < 16; ++j) {
        const int kc = k0 + scol + j;
        float v = (src[j] - in_mean[kc]) * in_rstd[kc] + 1e-4f;
        dst[j] = f2bf_bits(lrelu(v));
      }
      if (k0 + KSTEP < K) __builtin_prefetch((const void*)(src + KSTEP), 0, 0);
    } else {
      const unsigned short* src =
          (const unsigned short*)INp + (size_t)(nBase + srow) * K + k0 + scol;
#if HAVE_ASYNC
      async_copy_b128(src, dst);
      async_copy_b128(src + 8, dst + 8);
#else
      *(uint4*)dst       = *(const uint4*)src;
      *(uint4*)(dst + 8) = *(const uint4*)(src + 8);
      if (k0 + KSTEP < K) __builtin_prefetch((const void*)(src + KSTEP), 0, 0);
#endif
    }
  };

  const int wave = tid >> 5;
  const int lane = tid & 31;
  const int m0   = (wave & 3) * 32;   // 2 M tiles: m0, m0+16
  const int n0   = (wave >> 2) * 64;  // 4 N tiles: n0 + 16*j
  const int fr   = lane & 15;
  const int kh   = (lane >> 4) * 16;  // lanes 0-15: K 0..15, lanes 16-31: K 16..31

  v8f acc[2][4] = {};

  stageA(0, 0);
  stageB(0, 0);

  const int nK = K / KSTEP;
  for (int ki = 0; ki < nK; ++ki) {
    const int buf = ki & 1;
    wait_async();
    __syncthreads();
    if (ki + 1 < nK) { stageA((ki + 1) * KSTEP, buf ^ 1); stageB((ki + 1) * KSTEP, buf ^ 1); }

    v16bf a0 = ldfrag(&ldsA[buf][(m0 + fr)      * KSTEP + kh]);
    v16bf a1 = ldfrag(&ldsA[buf][(m0 + 16 + fr) * KSTEP + kh]);
    v16bf b0 = ldfrag(&ldsB[buf][(n0 + fr)      * KSTEP + kh]);
    v16bf b1 = ldfrag(&ldsB[buf][(n0 + 16 + fr) * KSTEP + kh]);
    v16bf b2 = ldfrag(&ldsB[buf][(n0 + 32 + fr) * KSTEP + kh]);
    v16bf b3 = ldfrag(&ldsB[buf][(n0 + 48 + fr) * KSTEP + kh]);

    acc[0][0] = bfwmma(a0, b0, acc[0][0]);
    acc[1][0] = bfwmma(a1, b0, acc[1][0]);
    acc[0][1] = bfwmma(a0, b1, acc[0][1]);
    acc[1][1] = bfwmma(a1, b1, acc[1][1]);
    acc[0][2] = bfwmma(a0, b2, acc[0][2]);
    acc[1][2] = bfwmma(a1, b2, acc[1][2]);
    acc[0][3] = bfwmma(a0, b3, acc[0][3]);
    acc[1][3] = bfwmma(a1, b3, acc[1][3]);
  }

  // ---- epilogue: C layout VGPR r -> M = r (lanes<16) / 8+r (lanes>=16) ----
  const int colL  = lane & 15;
  const int hiAdd = (lane >> 4) * 8;
#pragma unroll
  for (int mi = 0; mi < 2; ++mi) {
#pragma unroll
    for (int nj = 0; nj < 4; ++nj) {
#pragma unroll
      for (int rr = 0; rr < 8; ++rr) {
        const int   mLocal = m0 + mi * 16 + rr + hiAdd;
        const int   mg     = mBase + mLocal;
        const int   ng     = nBase + n0 + nj * 16 + colL;
        const float v      = acc[mi][nj][rr];
        if (OUT_BF16) {
          ((unsigned short*)OUTp)[(size_t)ng * out_stride + out_off + mg] =
              f2bf_bits(lrelu(v));
        } else {
          ((float*)OUTp)[(size_t)ng * out_stride + mg] = v;
          atomicAdd(&lsum[mLocal], v);
          atomicAdd(&lsq[mLocal],  v * v);
        }
      }
    }
  }
  if (!OUT_BF16) {
    __syncthreads();
    if (tid < TILE_M) {
      atomicAdd(&stat_sum[mBase + tid], lsum[tid]);
      atomicAdd(&stat_sq[mBase + tid],  lsq[tid]);
    }
  }
}

// ---------------------------------------------------------------------------
// Small prep / elementwise kernels
// ---------------------------------------------------------------------------
__global__ void zero_f32(float* p, int n) {
  for (int i = blockIdx.x * blockDim.x + threadIdx.x; i < n;
       i += gridDim.x * blockDim.x)
    p[i] = 0.f;
}

__global__ void f32_to_bf16(const float* __restrict__ s,
                            unsigned short* __restrict__ d, int n) {
  int i = blockIdx.x * blockDim.x + threadIdx.x;
  if (i < n) d[i] = f2bf_bits(s[i]);
}

// dconv_w [15][128][128][3] -> A [15][o=128][tap*128 + i] bf16
__global__ void prep_dconv(const float* __restrict__ dw,
                           unsigned short* __restrict__ A) {
  const int ho = blockIdx.x;            // 15*128
  const int h  = ho / DD;
  const int o  = ho % DD;
  const int i  = threadIdx.x;           // 128
#pragma unroll
  for (int k = 0; k < 3; ++k) {
    float v = dw[(((size_t)h * DD + o) * DD + i) * 3 + k];
    A[((size_t)h * DD + o) * 384 + k * DD + i] = f2bf_bits(v);
  }
}

// fofe_linear on the query + Linear + ReLU -> qb bf16 [B][128]
__global__ void qcode_kernel(const float* __restrict__ q_emb,
                             const float* __restrict__ qlin,
                             unsigned short* __restrict__ qb) {
  __shared__ float ws[DD];
  const int b = blockIdx.x, d = threadIdx.x;
  float s = 0.f, w = 1.f;
  for (int q = LQ - 1; q >= 0; --q) {
    s += w * q_emb[((size_t)b * LQ + q) * DD + d];
    w *= ALPHAF;
  }
  ws[d] = s;
  __syncthreads();
  float o = 0.f;
  for (int k = 0; k < DD; ++k) o += qlin[(size_t)d * DD + k] * ws[k];
  qb[b * DD + d] = f2bf_bits(fmaxf(o, 0.f));
}

// depthwise FOFE filter per window length -> y0pad bf16 [(h*B+b)*ROWS + row][128]
__global__ void fofe_kernel(const float* __restrict__ doc_emb,
                            unsigned short* __restrict__ y0) {
  const int idx = blockIdx.x;           // (h*B + b)*ROWS + row
  const int row = idx % ROWS;
  const int hb  = idx / ROWS;
  const int b   = hb % BB;
  const int h   = hb / BB;
  const int d   = threadIdx.x;
  const int len = h + 2;
  float v = 0.f;
  if (row >= 16 && row < 16 + LL) {
    const int l   = row - 16;
    const int pad = (len - 1) >> 1;
    float w = 1.f;                      // filt[len-1] = alpha^0
    for (int j = len - 1; j >= 0; --j) {
      const int p = l - pad + j;
      if (p >= 0 && p < LL) v += w * doc_emb[((size_t)b * LL + p) * DD + d];
      w *= ALPHAF;
    }
  }
  y0[(size_t)idx * DD + d] = f2bf_bits(v);
}

// X0[pix][tap*128 + i] = y0pad row (16 + l + (tap-1)*len)
__global__ void gather_kernel(const unsigned short* __restrict__ y0,
                              unsigned short* __restrict__ X0) {
  const int p  = blockIdx.x;            // (h*B + b)*L + l
  const int l  = p % LL;
  const int hb = p / LL;
  const int h  = hb / BB;
  const int len = h + 2;
  const int d  = threadIdx.x;
#pragma unroll
  for (int k = 0; k < 3; ++k) {
    const int row = 16 + l + (k - 1) * len;
    X0[(size_t)p * 384 + k * DD + d] =
        y0[((size_t)hb * ROWS + row) * DD + d];
  }
}

// X channels 128..255 = q_code broadcast
__global__ void qfill_kernel(const unsigned short* __restrict__ qb,
                             unsigned short* __restrict__ X) {
  const int p = blockIdx.x;
  const int b = (p / LL) % BB;
  const int d = threadIdx.x;
  X[(size_t)p * 256 + 128 + d] = qb[b * DD + d];
}

__global__ void bn_finalize(const float* __restrict__ sum,
                            const float* __restrict__ sq,
                            float* __restrict__ mean,
                            float* __restrict__ rstd, int C, float invN) {
  const int c = blockIdx.x * blockDim.x + threadIdx.x;
  if (c < C) {
    const float m   = sum[c] * invN;
    const float var = sq[c] * invN - m * m;
    mean[c] = m;
    rstd[c] = rsqrtf(var + 1e-5f);
  }
}

// s/e heads: per (b,l) contract lrelu(bn(h3)) with s_conv_w/e_conv_w over (c,h)
__global__ void __launch_bounds__(256)
head_kernel(const float* __restrict__ h3, const float* __restrict__ mean3,
            const float* __restrict__ rstd3, const float* __restrict__ sw,
            const float* __restrict__ ew, float* __restrict__ sbuf,
            float* __restrict__ ebuf) {
  __shared__ float rs[256], re[256];
  const int b = blockIdx.x >> 10;
  const int l = blockIdx.x & 1023;
  const int c = threadIdx.x;
  const float m = mean3[c], r = rstd3[c];
  float as = 0.f, ae = 0.f;
  for (int h = 0; h < NH; ++h) {
    const size_t p = ((size_t)(h * BB + b)) * LL + l;
    const float  a = lrelu((h3[p * 256 + c] - m) * r + 1e-4f);
    as += a * sw[c * NH + h];
    ae += a * ew[c * NH + h];
  }
  rs[c] = as; re[c] = ae;
  __syncthreads();
  for (int st = 128; st > 0; st >>= 1) {
    if (c < st) { rs[c] += rs[c + st]; re[c] += re[c + st]; }
    __syncthreads();
  }
  if (c == 0) { sbuf[b * LL + l] = rs[0]; ebuf[b * LL + l] = re[0]; }
}

// masked log-softmax over L; block per (which,b); out = [s(8x1024), e(8x1024)]
__global__ void __launch_bounds__(256)
logsoftmax_kernel(const float* __restrict__ sbuf, const float* __restrict__ ebuf,
                  const unsigned char* __restrict__ mask, float* __restrict__ out) {
  __shared__ float red[256];
  const int which = blockIdx.x >> 3;
  const int b     = blockIdx.x & 7;
  const float* src = which ? ebuf : sbuf;
  const int t = threadIdx.x;
  float vals[4];
  float mx = -INFINITY;
#pragma unroll
  for (int j = 0; j < 4; ++j) {
    const int l = t * 4 + j;
    float v = src[b * LL + l];
    if (mask[b * LL + l]) v = -INFINITY;
    vals[j] = v;
    mx = fmaxf(mx, v);
  }
  red[t] = mx; __syncthreads();
  for (int st = 128; st > 0; st >>= 1) {
    if (t < st) red[t] = fmaxf(red[t], red[t + st]);
    __syncthreads();
  }
  const float gmax = red[0];
  __syncthreads();
  float se = 0.f;
#pragma unroll
  for (int j = 0; j < 4; ++j) se += __expf(vals[j] - gmax);
  red[t] = se; __syncthreads();
  for (int st = 128; st > 0; st >>= 1) {
    if (t < st) red[t] += red[t + st];
    __syncthreads();
  }
  const float lse = logf(red[0]);
#pragma unroll
  for (int j = 0; j < 4; ++j)
    out[((size_t)which * BB + b) * LL + t * 4 + j] = vals[j] - gmax - lse;
}

// ---------------------------------------------------------------------------
// Host-side orchestration
// ---------------------------------------------------------------------------
extern "C" void kernel_launch(void* const* d_in, const int* in_sizes, int n_in,
                              void* d_out, int out_size, void* d_ws, size_t ws_size,
                              hipStream_t stream) {
  (void)in_sizes; (void)n_in; (void)out_size; (void)ws_size;

  const float* query_emb = (const float*)d_in[0];
  const float* doc_emb   = (const float*)d_in[2];
  const unsigned char* doc_mask = (const unsigned char*)d_in[3];
  const float* dconv_w   = (const float*)d_in[4];
  const float* q_lin_w   = (const float*)d_in[5];
  const float* fnn_w1    = (const float*)d_in[6];
  const float* fnn_w2    = (const float*)d_in[7];
  const float* fnn_w3    = (const float*)d_in[8];
  const float* s_conv_w  = (const float*)d_in[9];
  const float* e_conv_w  = (const float*)d_in[10];
  float* out_f = (float*)d_out;

  // ---- workspace carving (256B aligned) ----
  char* base = (char*)d_ws;
  size_t off = 0;
  auto carve = [&](size_t bytes) -> char* {
    char* p = base + off;
    off = (off + bytes + 255) & ~(size_t)255;
    return p;
  };
  unsigned short* qb   = (unsigned short*)carve((size_t)BB * DD * 2);
  unsigned short* y0   = (unsigned short*)carve((size_t)NH * BB * ROWS * DD * 2);
  unsigned short* X0   = (unsigned short*)carve((size_t)NPIX * 384 * 2);
  unsigned short* X    = (unsigned short*)carve((size_t)NPIX * 256 * 2);
  unsigned short* Adc  = (unsigned short*)carve((size_t)NH * DD * 384 * 2);
  unsigned short* W1b  = (unsigned short*)carve((size_t)512 * 256 * 2);
  unsigned short* W2b  = (unsigned short*)carve((size_t)512 * 512 * 2);
  unsigned short* W3b  = (unsigned short*)carve((size_t)256 * 512 * 2);
  float* h1    = (float*)carve((size_t)NPIX * 512 * 4);   // reused as h3
  float* h2    = (float*)carve((size_t)NPIX * 512 * 4);
  float* stats = (float*)carve((size_t)(512 * 8 + 256 * 4) * 4);
  float* sum1 = stats, *sq1 = sum1 + 512, *mean1 = sq1 + 512, *rstd1 = mean1 + 512;
  float* sum2 = rstd1 + 512, *sq2 = sum2 + 512, *mean2 = sq2 + 512, *rstd2 = mean2 + 512;
  float* sum3 = rstd2 + 512, *sq3 = sum3 + 256, *mean3 = sq3 + 256, *rstd3 = mean3 + 256;
  float* sbuf = (float*)carve((size_t)BB * LL * 4);
  float* ebuf = (float*)carve((size_t)BB * LL * 4);
  float* h3 = h1;  // layer3 reads h2, h1 is dead by then

  const int nStats = 512 * 8 + 256 * 4;
  zero_f32<<<8, 256, 0, stream>>>(stats, nStats);

  // weight conversions
  f32_to_bf16<<<(512 * 256 + 255) / 256, 256, 0, stream>>>(fnn_w1, W1b, 512 * 256);
  f32_to_bf16<<<(512 * 512 + 255) / 256, 256, 0, stream>>>(fnn_w2, W2b, 512 * 512);
  f32_to_bf16<<<(256 * 512 + 255) / 256, 256, 0, stream>>>(fnn_w3, W3b, 256 * 512);
  prep_dconv<<<NH * DD, DD, 0, stream>>>(dconv_w, Adc);

  // query code + FOFE + gather + q broadcast
  qcode_kernel<<<BB, DD, 0, stream>>>(query_emb, q_lin_w, qb);
  fofe_kernel<<<NH * BB * ROWS, DD, 0, stream>>>(doc_emb, y0);
  gather_kernel<<<NPIX, DD, 0, stream>>>(y0, X0);
  qfill_kernel<<<NPIX, DD, 0, stream>>>(qb, X);

  // dilated 3-tap conv as 128x384xN GEMM per window length, lrelu -> X[:,0:128]
  for (int h = 0; h < NH; ++h) {
    const int Nh = BB * LL;  // 8192
    gemm_wmma<0, 1><<<dim3(Nh / TILE_N, DD / TILE_M), 256, 0, stream>>>(
        Adc + (size_t)h * DD * 384,
        (const void*)(X0 + (size_t)h * Nh * 384),
        nullptr, nullptr,
        (void*)(X + (size_t)h * Nh * 256),
        nullptr, nullptr,
        DD, Nh, 384, 256, 0);
  }

  const float invN = 1.0f / (float)NPIX;

  // layer 1: 512 x 256 x NPIX  (bf16 in, f32 out + stats)
  gemm_wmma<0, 0><<<dim3(NPIX / TILE_N, 512 / TILE_M), 256, 0, stream>>>(
      W1b, (const void*)X, nullptr, nullptr, (void*)h1, sum1, sq1,
      512, NPIX, 256, 512, 0);
  bn_finalize<<<2, 256, 0, stream>>>(sum1, sq1, mean1, rstd1, 512, invN);

  // layer 2: 512 x 512 x NPIX  (BN+lrelu fused on B load)
  gemm_wmma<1, 0><<<dim3(NPIX / TILE_N, 512 / TILE_M), 256, 0, stream>>>(
      W2b, (const void*)h1, mean1, rstd1, (void*)h2, sum2, sq2,
      512, NPIX, 512, 512, 0);
  bn_finalize<<<2, 256, 0, stream>>>(sum2, sq2, mean2, rstd2, 512, invN);

  // layer 3: 256 x 512 x NPIX
  gemm_wmma<1, 0><<<dim3(NPIX / TILE_N, 256 / TILE_M), 256, 0, stream>>>(
      W3b, (const void*)h2, mean2, rstd2, (void*)h3, sum3, sq3,
      256, NPIX, 512, 256, 0);
  bn_finalize<<<1, 256, 0, stream>>>(sum3, sq3, mean3, rstd3, 256, invN);

  // heads + log-softmax
  head_kernel<<<BB * LL, 256, 0, stream>>>(h3, mean3, rstd3, s_conv_w, e_conv_w,
                                           sbuf, ebuf);
  logsoftmax_kernel<<<16, 256, 0, stream>>>(sbuf, ebuf, doc_mask, out_f);
}
